// TinyGRU_34119220199509
// MI455X (gfx1250) — matrix operand order repaired and checked by reading, not compiled, not run on hardware
//
#include <hip/hip_runtime.h>
#include <hip/hip_bf16.h>

// Problem dims (fixed by the reference)
#define B_  512
#define S_  1024
#define I_  64
#define H_  128
#define G_  384     // 3*H
#define C_  10
#define BT  16      // batch rows per workgroup -> 32 persistent workgroups
#define NTHREADS 256
#define TCH 32      // timesteps per prep-kernel block

typedef __attribute__((ext_vector_type(16))) __bf16          bf16x16;
typedef __attribute__((ext_vector_type(16))) unsigned short  u16x16;
typedef __attribute__((ext_vector_type(8)))  float           f32x8;

static __device__ __forceinline__ unsigned short f2bf(float f) {
  union { float f; unsigned u; } c; c.f = f;
  unsigned u = c.u + 0x7FFFu + ((c.u >> 16) & 1u);   // round-to-nearest-even
  return (unsigned short)(u >> 16);
}

static __device__ __forceinline__ f32x8 wmma_bf16(u16x16 a, u16x16 b, f32x8 c) {
  return __builtin_amdgcn_wmma_f32_16x16x32_bf16(
      false, __builtin_bit_cast(bf16x16, a),
      false, __builtin_bit_cast(bf16x16, b),
      (short)0, c, false, false);
}

static __device__ __forceinline__ float fast_rcp(float x) {
#if __has_builtin(__builtin_amdgcn_rcpf)
  return __builtin_amdgcn_rcpf(x);        // single v_rcp_f32
#else
  return 1.0f / x;
#endif
}
static __device__ __forceinline__ float sigmoid_(float x) {
  return fast_rcp(1.0f + __expf(-x));
}
static __device__ __forceinline__ float tanh_(float x) {
#if __has_builtin(__builtin_amdgcn_tanhf)
  return __builtin_amdgcn_tanhf(x);       // v_tanh_f32 (gfx1250 TRANS op)
#elif __has_builtin(__builtin_amdgcn_tanh_f32)
  return __builtin_amdgcn_tanh_f32(x);
#else
  float a = fabsf(x);
  float e = __expf(-2.0f * a);
  float t = (1.0f - e) * fast_rcp(1.0f + e);
  return copysignf(t, x);
#endif
}

// ---------------------------------------------------------------------------
// Kernel 1 (parallel, throughput-bound): xg0 = x @ W_ih_l0^T for ALL (b, t),
// written pre-swizzled so the recurrent kernel's gate stage reads 96
// contiguous bytes per lane:  ws[((bt*S + t)*8 + w)*32 + lane][j(3)][e(8)] f32
// ---------------------------------------------------------------------------
__global__ __launch_bounds__(NTHREADS, 1) void gru_xg0_prep(
    const float* __restrict__ x, const float* __restrict__ Wih0,
    float* __restrict__ ws)
{
  const int tid  = threadIdx.x;
  const int lane = tid & 31;
  const int w    = tid >> 5;
  const int hi   = lane >> 4;
  const int cj   = lane & 15;
  const int bt   = blockIdx.x;            // batch tile 0..31
  const int t0   = blockIdx.y * TCH;      // timestep chunk
  const int bbase = bt * BT;
  const int kbase = hi << 4;

  u16x16 wih0f[2][3];
  for (int kt = 0; kt < 2; ++kt)
    for (int j = 0; j < 3; ++j) {
      int n = (w + 8 * j) * 16 + cj;
      u16x16 v;
      #pragma unroll
      for (int e = 0; e < 16; ++e) v[e] = f2bf(Wih0[n * I_ + kt * 32 + 8 * hi + (e & 7) + ((e >> 3) << 4)]);
      wih0f[kt][j] = v;
    }

  const f32x8 z8 = {0.f, 0.f, 0.f, 0.f, 0.f, 0.f, 0.f, 0.f};
  for (int tt = 0; tt < TCH; ++tt) {
    const int t = t0 + tt;
    f32x8 aR = z8, aZ = z8, aN = z8;
    #pragma unroll
    for (int kt = 0; kt < 2; ++kt) {
      const float* xrow = x + ((size_t)(bbase + cj) * S_ + t) * I_ + kt * 32 + 8 * hi;
      u16x16 xa;                     // 16-bit A layout: K = 8*hi + e (+8 for e>=8)
      #pragma unroll
      for (int e = 0; e < 8; ++e) xa[e]     = f2bf(xrow[e]);
      #pragma unroll
      for (int e = 0; e < 8; ++e) xa[e + 8] = f2bf(xrow[16 + e]);
      aR = wmma_bf16(xa, wih0f[kt][0], aR);
      aZ = wmma_bf16(xa, wih0f[kt][1], aZ);
      aN = wmma_bf16(xa, wih0f[kt][2], aN);
    }
    float* op = ws + ((((size_t)bt * S_ + t) * 8 + w) * 32 + lane) * 24;
    #pragma unroll
    for (int e = 0; e < 8; ++e) { op[e] = aR[e]; op[8 + e] = aZ[e]; op[16 + e] = aN[e]; }
  }
}

// ---------------------------------------------------------------------------
// Kernel 2 (persistent, latency-bound): fused 2-layer GRU recurrence + FC.
// Double-buffered h fragments -> exactly ONE __syncthreads per timestep.
// ---------------------------------------------------------------------------
#define WHH0_OFF  0u        // 96 KB: W_hh_l0 bf16 B-fragments (4 kt x 24 nt x 1KB)
#define HA0_OFF   98304u    //  8 KB: h0 A-fragments, double buffered (2 x 4KB)
#define HA1_OFF   106496u   //  8 KB: h1 A-fragments, double buffered
#define XG_OFF    114688u   // 48 KB: xg0 slabs, double buffer x 8 waves x 3KB
#define HFIN_OFF  114688u   //  8 KB: final h1 (overlays XG after the loop)
#define SMEM_BYTES 163840u  // 160 KB total

__global__ __launch_bounds__(NTHREADS, 1) void tiny_gru_fused(
    const float* __restrict__ xg0ws,
    const float* __restrict__ Whh0,
    const float* __restrict__ bih0, const float* __restrict__ bhh0,
    const float* __restrict__ Wih1, const float* __restrict__ Whh1,
    const float* __restrict__ bih1, const float* __restrict__ bhh1,
    const float* __restrict__ fcw,  const float* __restrict__ fcb,
    float* __restrict__ out)
{
  __shared__ __attribute__((aligned(32))) unsigned char smem[SMEM_BYTES];
  unsigned short* s_u16 = reinterpret_cast<unsigned short*>(smem);

  const int tid   = threadIdx.x;
  const int lane  = tid & 31;
  const int w     = tid >> 5;        // wave id 0..7: owns h columns [16w,16w+16)
  const int hi    = lane >> 4;
  const int cj    = lane & 15;
  const int bt    = blockIdx.x;
  const int bbase = bt * BT;
  const int kbase = hi << 4;

  // ---- one-time: W_hh_l0 -> LDS B-fragments (bf16) ----
  for (int idx = tid; idx < 96 * 512; idx += NTHREADS) {
    int frag = idx >> 9, r = idx & 511;
    int la = r >> 4, e = r & 15;
    int kt = frag / 24, nt = frag % 24;
    int n = nt * 16 + (la & 15);
    int k = kt * 32 + ((la >> 4) << 4) + e;
    s_u16[idx] = f2bf(Whh0[n * H_ + k]);
  }
  // zero h fragments (both parities of hA0 + hA1: 16 KB = 8192 ushorts)
  for (int idx = tid; idx < 8192; idx += NTHREADS)
    reinterpret_cast<unsigned short*>(smem + HA0_OFF)[idx] = 0;

  // ---- one-time: layer-1 weight B-fragments in registers ----
  u16x16 wih1f[4][3], whh1f[4][3];
  for (int kt = 0; kt < 4; ++kt)
    for (int j = 0; j < 3; ++j) {
      int n = (w + 8 * j) * 16 + cj;
      u16x16 v, v2;
      #pragma unroll
      for (int e = 0; e < 16; ++e) v[e]  = f2bf(Wih1[n * H_ + kt * 32 + kbase + e]);
      #pragma unroll
      for (int e = 0; e < 16; ++e) v2[e] = f2bf(Whh1[n * H_ + kt * 32 + kbase + e]);
      wih1f[kt][j] = v; whh1f[kt][j] = v2;
    }

  // Per-lane gate biases
  const float brz0_r = bih0[w * 16 + cj]       + bhh0[w * 16 + cj];
  const float brz0_z = bih0[(w + 8) * 16 + cj] + bhh0[(w + 8) * 16 + cj];
  const float bn0_i  = bih0[(w + 16) * 16 + cj];
  const float bn0_h  = bhh0[(w + 16) * 16 + cj];
  const float brz1_r = bih1[w * 16 + cj]       + bhh1[w * 16 + cj];
  const float brz1_z = bih1[(w + 8) * 16 + cj] + bhh1[(w + 8) * 16 + cj];
  const float bn1_i  = bih1[(w + 16) * 16 + cj];
  const float bn1_h  = bhh1[(w + 16) * 16 + cj];

  // A-fragment scatter constants (inverse of 16-bit A layout)
  const int colk  = ((w & 1) << 4) + cj;
  const int ktH   = w >> 1;
  const int gA    = colk >> 4;
  const int eA    = colk & 15;
  const int wbA   = ktH * 512 + eA;

  // Async xg0 staging: wave-private 3KB slabs, double buffered.
  const float* xg_wave_base = xg0ws + ((size_t)bt * S_ * 8 + w) * 768; // +t*8*768
  const unsigned xg_lds_w   = XG_OFF + (unsigned)w * 3072;             // + d*8*3072

  auto stage_xg = [&](int t) {            // issue 6 async b128 ops for step t
    const char* g = (const char*)(xg_wave_base + (size_t)t * 6144) + lane * 16;
    unsigned l = xg_lds_w + (unsigned)(t & 1) * 24576u + (unsigned)lane * 16u;
    #pragma unroll
    for (int i = 0; i < 6; ++i) {
      asm volatile("global_load_async_to_lds_b128 %0, %1, off"
                   :: "v"(l + (unsigned)(i * 512)), "v"(g + i * 512)
                   : "memory");
    }
  };

  const f32x8 z8 = {0.f, 0.f, 0.f, 0.f, 0.f, 0.f, 0.f, 0.f};
  f32x8 h0 = z8, h1 = z8;

  stage_xg(0);                 // prologue: stage t=0 into buffer 0
  __syncthreads();

  for (int t = 0; t < S_; ++t) {
    const unsigned pr = (unsigned)(t & 1);       // read parity
    const unsigned pw = pr ^ 1u;                 // write parity
    if (t + 1 < S_) stage_xg(t + 1);             // overlap next slab

    // ---- layer0 recurrent GEMM: gh0 = h0 @ Whh0^T ----
    f32x8 accR = z8, accZ = z8, accHN = z8;
    #pragma unroll
    for (int kt = 0; kt < 4; ++kt) {
      u16x16 ha = *reinterpret_cast<const u16x16*>(
          &s_u16[HA0_OFF / 2 + pr * 2048 + kt * 512 + lane * 16]);
      u16x16 w0 = *reinterpret_cast<const u16x16*>(&s_u16[(kt * 24 + w)      * 512 + lane * 16]);
      u16x16 w1 = *reinterpret_cast<const u16x16*>(&s_u16[(kt * 24 + w + 8)  * 512 + lane * 16]);
      u16x16 w2 = *reinterpret_cast<const u16x16*>(&s_u16[(kt * 24 + w + 16) * 512 + lane * 16]);
      accR  = wmma_bf16(ha, w0, accR);
      accZ  = wmma_bf16(ha, w1, accZ);
      accHN = wmma_bf16(ha, w2, accHN);
    }
    // this step's xg slab: oldest 6 of at most 12 outstanding async loads
    if (t + 1 < S_) { asm volatile("s_wait_asynccnt 0x6" ::: "memory"); }
    else            { asm volatile("s_wait_asynccnt 0x0" ::: "memory"); }

    // ---- layer0 gates -> h0; scatter new h0 A-fragments (write parity) ----
    const float* xgl = reinterpret_cast<const float*>(
        smem + xg_lds_w + pr * 24576u + (unsigned)lane * 96u);
    #pragma unroll
    for (int e = 0; e < 8; ++e) {
      float r  = sigmoid_(accR[e] + xgl[e]      + brz0_r);
      float z  = sigmoid_(accZ[e] + xgl[8 + e]  + brz0_z);
      float n  = tanh_((xgl[16 + e] + bn0_i) + r * (accHN[e] + bn0_h));
      float hn = (1.0f - z) * n + z * h0[e];
      h0[e] = hn;
      int m = e + 8 * hi;
      s_u16[HA0_OFF / 2 + pw * 2048 + wbA + (m + 16 * gA) * 16] = f2bf(hn);
    }
    __syncthreads();   // the ONLY barrier: new h0 frags -> layer1 GEMMs

    // ---- layer1: xg1 (from new h0, write parity) + gh1 (old h1, read parity)
    f32x8 bR = z8, bZ = z8, bIN = z8, bHN = z8;
    #pragma unroll
    for (int kt = 0; kt < 4; ++kt) {
      u16x16 a0 = *reinterpret_cast<const u16x16*>(
          &s_u16[HA0_OFF / 2 + pw * 2048 + kt * 512 + lane * 16]);
      bR  = wmma_bf16(a0, wih1f[kt][0], bR);
      bZ  = wmma_bf16(a0, wih1f[kt][1], bZ);
      bIN = wmma_bf16(a0, wih1f[kt][2], bIN);
      u16x16 a1 = *reinterpret_cast<const u16x16*>(
          &s_u16[HA1_OFF / 2 + pr * 2048 + kt * 512 + lane * 16]);
      bR  = wmma_bf16(a1, whh1f[kt][0], bR);
      bZ  = wmma_bf16(a1, whh1f[kt][1], bZ);
      bHN = wmma_bf16(a1, whh1f[kt][2], bHN);
    }

    // ---- layer1 gates -> h1; scatter new h1 A-fragments (write parity) ----
    #pragma unroll
    for (int e = 0; e < 8; ++e) {
      float r  = sigmoid_(bR[e] + brz1_r);
      float z  = sigmoid_(bZ[e] + brz1_z);
      float n  = tanh_((bIN[e] + bn1_i) + r * (bHN[e] + bn1_h));
      float hn = (1.0f - z) * n + z * h1[e];
      h1[e] = hn;
      int m = e + 8 * hi;
      s_u16[HA1_OFF / 2 + pw * 2048 + wbA + (m + 16 * gA) * 16] = f2bf(hn);
    }

    // prefetch the t+2 slab toward cache (global_prefetch_b8)
    if (t + 2 < S_)
      __builtin_prefetch((const char*)(xg_wave_base + (size_t)(t + 2) * 6144) + lane * 16, 0, 0);
  }

  // ---- FC head on final h1 (reuse XG region as f32 [16][128]) ----
  __syncthreads();
  float* hfin = reinterpret_cast<float*>(smem + HFIN_OFF);
  #pragma unroll
  for (int e = 0; e < 8; ++e) {
    int m = e + 8 * hi;
    hfin[m * H_ + w * 16 + cj] = h1[e];
  }
  __syncthreads();
  for (int o = tid; o < BT * C_; o += NTHREADS) {
    int m = o / C_, c = o % C_;
    float acc = fcb[c];
    #pragma unroll 8
    for (int k = 0; k < H_; ++k) acc += hfin[m * H_ + k] * fcw[c * H_ + k];
    out[(size_t)(bbase + m) * C_ + c] = acc;
  }
}

extern "C" void kernel_launch(void* const* d_in, const int* in_sizes, int n_in,
                              void* d_out, int out_size, void* d_ws, size_t ws_size,
                              hipStream_t stream) {
  (void)in_sizes; (void)n_in; (void)ws_size; (void)out_size;
  const float* x    = (const float*)d_in[0];
  const float* Wih0 = (const float*)d_in[1];
  const float* Whh0 = (const float*)d_in[2];
  const float* bih0 = (const float*)d_in[3];
  const float* bhh0 = (const float*)d_in[4];
  const float* Wih1 = (const float*)d_in[5];
  const float* Whh1 = (const float*)d_in[6];
  const float* bih1 = (const float*)d_in[7];
  const float* bhh1 = (const float*)d_in[8];
  const float* fcw  = (const float*)d_in[9];
  const float* fcb  = (const float*)d_in[10];
  float* out   = (float*)d_out;
  float* xg0ws = (float*)d_ws;   // needs B*S*3H*4 = 805,306,368 bytes

  gru_xg0_prep<<<dim3(B_ / BT, S_ / TCH), dim3(NTHREADS), 0, stream>>>(x, Wih0, xg0ws);
  tiny_gru_fused<<<dim3(B_ / BT), dim3(NTHREADS), 0, stream>>>(
      xg0ws, Whh0, bih0, bhh0, Wih1, Whh1, bih1, bhh1, fcw, fcb, out);
}
